// LRPDenseNetTransition_21655225106491
// MI455X (gfx1250) — matrix operand order repaired and checked by reading, not compiled, not run on hardware
//
#include <hip/hip_runtime.h>

typedef __attribute__((ext_vector_type(16))) __bf16 v16bf;
typedef __attribute__((ext_vector_type(8)))  __bf16 v8bf;
typedef __attribute__((ext_vector_type(8)))  float  v8f;

#define KK   7
#define BB   64
#define CC   128
#define HP   14
#define HH   28
#define CIN  256
#define EPS  0.01f
#define BNE  1e-5f

__device__ __forceinline__ float stab(float z) {
    return z + (z >= 0.0f ? EPS : -EPS);
}

// ---------------------------------------------------------------------------
// Kernel 1: fused elementwise LRP chain.
// One thread per (b, c, y, x) pool cell: accumulates the K=7 BN z-rule terms,
// forms s_pool, then scatters the 2x2 upsample with the ReLU gate and the
// epsilon-stabilized divide by aK_conv. Output is scT[b][hw][c] in bf16 so the
// GEMM kernel sees a row-major [M=50176, K=128] A matrix with contiguous K.
// ---------------------------------------------------------------------------
__global__ void lrp_elem_kernel(const float* __restrict__ rK,
                                const float* __restrict__ aK,
                                const float* __restrict__ gamma,
                                const float* __restrict__ var,
                                const float* __restrict__ aKPool,
                                const float* __restrict__ aJ_pool,
                                const float* __restrict__ aK_conv,
                                __bf16* __restrict__ scT) {
    const int idx = blockIdx.x * blockDim.x + threadIdx.x;   // (b,c,y,x)
    const int total = BB * CC * HP * HP;
    if (idx >= total) return;
    const int x  = idx % HP;
    const int y  = (idx / HP) % HP;
    const int ch = (idx / (HP * HP)) % CC;
    const int b  = idx / (HP * HP * CC);
    const size_t KS = (size_t)BB * CC * HP * HP;

    float ssum = 0.0f;
#pragma unroll
    for (int k = 0; k < KK; ++k) {
        const float w = gamma[k * CC + ch] * rsqrtf(var[k * CC + ch] + BNE);
        const float a = aK[(size_t)k * KS + idx];
        const float r = rK[(size_t)k * KS + idx];
        ssum += (r / stab(a)) * w;
    }
    const float apool = aKPool[idx];
    const float spool = (apool * ssum) / stab(apool) * 0.25f;

#pragma unroll
    for (int dy = 0; dy < 2; ++dy) {
#pragma unroll
        for (int dx = 0; dx < 2; ++dx) {
            const int h = 2 * y + dy, wd = 2 * x + dx;
            const size_t off = (((size_t)b * CC + ch) * HH + h) * HH + wd;
            const float R  = aJ_pool[off] * spool;
            const float sc = R / stab(aK_conv[off]);
            scT[((size_t)b * (HH * HH) + h * HH + wd) * CC + ch] = (__bf16)sc;
        }
    }
}

// ---------------------------------------------------------------------------
// Kernel 2: GEMM [50176 x 128] x [128 x 256] via v_wmma_f32_16x16x32_bf16,
// fused with the aJ_conv gate. W is staged transposed in LDS (WT[cin][k],
// 64 KB bf16) so each lane's B-column K-slice is contiguous. Each wave owns
// one 16-row M-tile (A held in registers, 4 K-chunks of 32) and sweeps all
// 16 N-tiles: 4 WMMAs each, f32 accumulate.
// ---------------------------------------------------------------------------
__global__ void __launch_bounds__(256)
lrp_gemm_kernel(const __bf16* __restrict__ scT,   // [50176][128] bf16
                const float*  __restrict__ W,     // [128][256]  f32
                const float*  __restrict__ aJ,    // [64][256][784]
                float*        __restrict__ out) { // [64][256][784]
    __shared__ __bf16 WT[CIN * CC];               // 64 KB

    const int tid = threadIdx.x;
    {   // transpose W into LDS as bf16: thread t owns output column c = t
        const int c = tid;                        // 0..255
        for (int k = 0; k < CC; ++k)
            WT[c * CC + k] = (__bf16)W[k * CIN + c];
    }
    __syncthreads();

    const int wave  = tid >> 5;
    const int lane  = tid & 31;
    const int lhalf = lane >> 4;                  // 0: lanes 0-15, 1: lanes 16-31
    const int l16   = lane & 15;

    const int m_tile = blockIdx.x * 8 + wave;     // 392 blocks * 8 waves = 3136
    const int m_base = m_tile * 16;
    const int b      = m_base / (HH * HH);        // 784 % 16 == 0: tile stays in one b
    const int hwb    = m_base % (HH * HH);

    union Vu { v16bf v; v8bf h[2]; };

    // A operand: row M = l16, K-halves per ISA 16-bit A 16x32 layout:
    //   elems 0-7  -> K = kc*32 + lhalf*8 + [0..7]
    //   elems 8-15 -> K = kc*32 + 16 + lhalf*8 + [0..7]
    Vu A[4];
    const __bf16* aptr = scT + (size_t)(m_base + l16) * CC;
#pragma unroll
    for (int kc = 0; kc < 4; ++kc) {
        const int koff = kc * 32 + lhalf * 8;
        A[kc].h[0] = *(const v8bf*)(aptr + koff);
        A[kc].h[1] = *(const v8bf*)(aptr + koff + 16);
    }

#pragma unroll 1
    for (int nt = 0; nt < 16; ++nt) {
        const int n = nt * 16 + l16;              // output channel (N)
        v8f acc = {};
#pragma unroll
        for (int kc = 0; kc < 4; ++kc) {
            // B operand: column N = l16, elems 0-7 -> K = kc*32 + lhalf*8 + [0..7], elems 8-15 -> K = kc*32 + 16 + lhalf*8 + [0..7]
            Vu Bm;
            const __bf16* bptr = WT + n * CC + kc * 32 + lhalf * 8;
            Bm.h[0] = *(const v8bf*)(bptr);
            Bm.h[1] = *(const v8bf*)(bptr + 16);
            acc = __builtin_amdgcn_wmma_f32_16x16x32_bf16(
                false, A[kc].v, false, Bm.v, (short)0, acc, false, false);
        }
        // D layout: VGPR v holds (M = v + lhalf*8, N = n) -> 8 consecutive hw slots
        const size_t base = ((size_t)b * CIN + n) * (HH * HH) + hwb + lhalf * 8;
        const float4 a0 = *(const float4*)(aJ + base);
        const float4 a1 = *(const float4*)(aJ + base + 4);
        float4 o0, o1;
        o0.x = acc[0] * a0.x; o0.y = acc[1] * a0.y;
        o0.z = acc[2] * a0.z; o0.w = acc[3] * a0.w;
        o1.x = acc[4] * a1.x; o1.y = acc[5] * a1.y;
        o1.z = acc[6] * a1.z; o1.w = acc[7] * a1.w;
        *(float4*)(out + base)     = o0;
        *(float4*)(out + base + 4) = o1;
    }
}

extern "C" void kernel_launch(void* const* d_in, const int* in_sizes, int n_in,
                              void* d_out, int out_size, void* d_ws, size_t ws_size,
                              hipStream_t stream) {
    (void)in_sizes; (void)n_in; (void)out_size; (void)ws_size;
    const float* rK      = (const float*)d_in[0];
    const float* aK      = (const float*)d_in[1];
    const float* gamma   = (const float*)d_in[2];
    const float* var     = (const float*)d_in[3];
    const float* aKPool  = (const float*)d_in[4];
    const float* aJ_pool = (const float*)d_in[5];
    const float* aK_conv = (const float*)d_in[6];
    const float* aJ_conv = (const float*)d_in[7];
    const float* W_conv  = (const float*)d_in[8];
    float* out = (float*)d_out;
    __bf16* scT = (__bf16*)d_ws;   // 50176 * 128 bf16 = 12.85 MB scratch

    const int total1 = BB * CC * HP * HP;          // 1,605,632 threads
    lrp_elem_kernel<<<(total1 + 255) / 256, 256, 0, stream>>>(
        rK, aK, gamma, var, aKPool, aJ_pool, aK_conv, scT);

    lrp_gemm_kernel<<<392, 256, 0, stream>>>(scT, W_conv, aJ_conv, out);
}
